// NAM_55963423867204
// MI455X (gfx1250) — matrix operand order repaired and checked by reading, not compiled, not run on hardware
//
#include <hip/hip_runtime.h>

typedef __attribute__((ext_vector_type(16))) _Float16 v16h;
typedef __attribute__((ext_vector_type(8)))  _Float16 v8h;
typedef __attribute__((ext_vector_type(8)))  float    v8f;
typedef __attribute__((ext_vector_type(4)))  float    v4f;

#define B_TOT 16384
#define FDIM  64
#define UDIM  64
#define H1DIM 64
#define H2DIM 32
#define BT    128   // batch rows per block
#define NW    8     // waves per block (256 threads, wave32)

// ---------------------------------------------------------------------------
// Kernel 1: per-(feature, 128-row batch tile) MLP via WMMA.
// One wave owns a 16-row M tile; all matmuls are v_wmma_f32_16x16x32_f16.
// ---------------------------------------------------------------------------
__global__ __launch_bounds__(256) void nam_forward(
    const float* __restrict__ inputs, const float* __restrict__ W0,
    const float* __restrict__ b0,     const float* __restrict__ W1,
    const float* __restrict__ b1,     const float* __restrict__ W2,
    const float* __restrict__ b2,     const float* __restrict__ Wo,
    const float* __restrict__ bo,
    float* __restrict__ y, float* __restrict__ drop)
{
    __shared__ __align__(32) _Float16 sW1T[H1DIM * UDIM];    // [h1][u]  (8 KB)
    __shared__ __align__(32) _Float16 sW2T[H2DIM * H1DIM];   // [h2][h1] (4 KB)
    __shared__ __align__(32) _Float16 sH1[NW * 16 * H1DIM];  // per-wave (16 KB)
    __shared__ __align__(32) _Float16 sH2[NW * 16 * H2DIM];  // per-wave (8 KB)
    __shared__ float    sExpW0[UDIM];
    __shared__ float    sB1[H1DIM];
    __shared__ float    sB2[H2DIM];
    __shared__ float    sIn[BT];
    __shared__ _Float16 sWo[H2DIM];
    __shared__ float    sB0f;

    const int t     = threadIdx.x;
    const int f     = blockIdx.y;
    const int bBase = blockIdx.x * BT;

    // ---- stage per-feature weights into LDS (f16, transposed for B-frags) ----
    const float* W1f = W1 + f * (UDIM * H1DIM);
#pragma unroll
    for (int i = 0; i < (UDIM * H1DIM) / 256; ++i) {
        int idx = i * 256 + t;                 // coalesced global read
        int u = idx / H1DIM, h = idx % H1DIM;
        sW1T[h * UDIM + u] = (_Float16)W1f[idx];
    }
    const float* W2f = W2 + f * (H1DIM * H2DIM);
#pragma unroll
    for (int i = 0; i < (H1DIM * H2DIM) / 256; ++i) {
        int idx = i * 256 + t;
        int a = idx / H2DIM, c = idx % H2DIM;
        sW2T[c * H1DIM + a] = (_Float16)W2f[idx];
    }
    if (t < UDIM)  sExpW0[t] = expf(W0[f * UDIM + t]);
    if (t < H1DIM) sB1[t] = b1[f * H1DIM + t];
    if (t < H2DIM) { sB2[t] = b2[f * H2DIM + t]; sWo[t] = (_Float16)Wo[f * H2DIM + t]; }
    if (t == 0)    sB0f = b0[f];
    if (t < BT)    sIn[t] = inputs[(bBase + t) * FDIM + f];   // column-f gather
    __syncthreads();

    const int wave = t >> 5;
    const int lane = t & 31;
    const int m    = lane & 15;   // M row (A/C) or N col (B/C) within tile
    const int hw   = lane >> 4;   // half-wave select
    const float bof = bo[f];

    // ---- layer 0: A-frags of h0 built in registers -------------------------
    // contiguous A k order used here (B matches): lane m holds row M=m; K=16*hw+e
    const float xb = sIn[wave * 16 + m] - sB0f;
    v16h A0[2];
#pragma unroll
    for (int k = 0; k < 2; ++k) {
#pragma unroll
        for (int e = 0; e < 16; ++e) {
            int u = 32 * k + 16 * hw + e;
            float v = fminf(fmaxf(xb * sExpW0[u], 0.f), 1.f);   // v_max/v_min
            A0[k][e] = (_Float16)v;
        }
    }

    _Float16* myH1 = sH1 + wave * (16 * H1DIM);
    _Float16* myH2 = sH2 + wave * (16 * H2DIM);

    // ---- layer 1: h1(16x64) = relu(h0(16x64) @ W1(64x64) + b1) -------------
#pragma unroll
    for (int n = 0; n < 4; ++n) {
        v8f acc = {};
        const int hcol = n * 16 + m;          // B-frag: lane column N = hcol
#pragma unroll
        for (int k = 0; k < 2; ++k) {
            // B layout: 16 contiguous K = 32k + 16*hw .. +15 for column hcol
            v16h Bf = *(const v16h*)&sW1T[hcol * UDIM + 32 * k + 16 * hw];
            acc = __builtin_amdgcn_wmma_f32_16x16x32_f16(
                      false, A0[k], false, Bf, (short)0, acc, false, false);
        }
        const float bb = sB1[n * 16 + m];
#pragma unroll
        for (int r = 0; r < 8; ++r) {         // C layout: row = r + 8*hw, col = m
            float v = fmaxf(acc[r] + bb, 0.f);
            myH1[(r + 8 * hw) * H1DIM + n * 16 + m] = (_Float16)v;
        }
    }

    // ---- layer 2: h2(16x32) = relu(h1 @ W2(64x32) + b2) --------------------
#pragma unroll
    for (int n = 0; n < 2; ++n) {
        v8f acc = {};
        const int hcol = n * 16 + m;
#pragma unroll
        for (int k = 0; k < 2; ++k) {
            v8h alo = *(const v8h*)&myH1[m * H1DIM + 32 * k + 16 * hw];
            v8h ahi = *(const v8h*)&myH1[m * H1DIM + 32 * k + 16 * hw + 8];
            v16h Af = __builtin_shufflevector(alo, ahi,
                        0,1,2,3,4,5,6,7,8,9,10,11,12,13,14,15);
            v16h Bf = *(const v16h*)&sW2T[hcol * H1DIM + 32 * k + 16 * hw];
            acc = __builtin_amdgcn_wmma_f32_16x16x32_f16(
                      false, Af, false, Bf, (short)0, acc, false, false);
        }
        const float bb = sB2[n * 16 + m];
#pragma unroll
        for (int r = 0; r < 8; ++r) {
            float v = fmaxf(acc[r] + bb, 0.f);
            myH2[(r + 8 * hw) * H2DIM + n * 16 + m] = (_Float16)v;
        }
    }

    // ---- layer 3: y(16) = h2(16x32) @ [Wo | 0...](32x16), answer in col 0 ---
    {
        v8h alo = *(const v8h*)&myH2[m * H2DIM + 8 * hw];
        v8h ahi = *(const v8h*)&myH2[m * H2DIM + 16 + 8 * hw];
        v16h Af = __builtin_shufflevector(alo, ahi,
                    0,1,2,3,4,5,6,7,8,9,10,11,12,13,14,15);
        v16h Bf;
#pragma unroll
        for (int e = 0; e < 16; ++e)
            Bf[e] = (m == 0) ? sWo[e < 8 ? 8 * hw + e : 16 + 8 * hw + (e - 8)] : (_Float16)0.f;
        v8f acc = {};
        acc = __builtin_amdgcn_wmma_f32_16x16x32_f16(
                  false, Af, false, Bf, (short)0, acc, false, false);
        if (m == 0) {                          // lanes 0 (rows 0-7) & 16 (rows 8-15)
#pragma unroll
            for (int r = 0; r < 8; ++r) {
                int brow = bBase + wave * 16 + r + 8 * hw;
                float v = acc[r] + bof;
                y[brow * FDIM + f] = v;                               // re-read by k2
                __builtin_nontemporal_store(v, &drop[brow * FDIM + f]); // write-only
            }
        }
    }
}

// ---------------------------------------------------------------------------
// Kernel 2: out[b] = sum_f y[b,f] + bias   (deterministic, no atomics)
// ---------------------------------------------------------------------------
__global__ __launch_bounds__(256) void nam_reduce(
    const float* __restrict__ y, const float* __restrict__ bias,
    float* __restrict__ out)
{
    int b = blockIdx.x * 256 + threadIdx.x;
    const v4f* row = (const v4f*)(y + b * FDIM);
    float s = 0.f;
#pragma unroll
    for (int i = 0; i < FDIM / 4; ++i) {
        v4f v = row[i];
        s += v[0] + v[1] + v[2] + v[3];
    }
    out[b] = s + bias[0];
}

extern "C" void kernel_launch(void* const* d_in, const int* in_sizes, int n_in,
                              void* d_out, int out_size, void* d_ws, size_t ws_size,
                              hipStream_t stream) {
    const float* inputs = (const float*)d_in[0];
    const float* W0     = (const float*)d_in[1];
    const float* b0     = (const float*)d_in[2];
    const float* W1     = (const float*)d_in[3];
    const float* b1     = (const float*)d_in[4];
    const float* W2     = (const float*)d_in[5];
    const float* b2     = (const float*)d_in[6];
    const float* Wo     = (const float*)d_in[7];
    const float* bo     = (const float*)d_in[8];
    const float* bias   = (const float*)d_in[9];

    float* out  = (float*)d_out;               // (B,1)
    float* drop = out + B_TOT;                 // (B,1,F)
    float* y    = drop + B_TOT * FDIM;         // (B,F)
    // interaction_effects_out has zero elements — nothing to write.

    dim3 grid(B_TOT / BT, FDIM);
    nam_forward<<<grid, 256, 0, stream>>>(inputs, W0, b0, W1, b1, W2, b2, Wo, bo,
                                          y, drop);
    nam_reduce<<<B_TOT / 256, 256, 0, stream>>>(y, bias, out);
}
